// AttentionSeqModel_7490422964470
// MI455X (gfx1250) — matrix operand order repaired and checked by reading, not compiled, not run on hardware
//
#include <hip/hip_runtime.h>
#include <hip/hip_bf16.h>

// ---------------------------------------------------------------------------
// AttentionSeqModel for MI455X (gfx1250, wave32, WMMA f32_16x16x32_f16)
// B_TILE=16 persistent recurrent kernels (64 WGs), parallel softmax/output.
// ---------------------------------------------------------------------------

typedef __attribute__((ext_vector_type(16))) _Float16 v16h;
typedef __attribute__((ext_vector_type(8)))  _Float16 v8h;
typedef __attribute__((ext_vector_type(8)))  float    v8f;

#define HID   128
#define SEQ   128
#define OBSZ  32
#define BATCH 1024
#define KFLAT 4096   // L * OBS

// ---- WMMA helpers ----------------------------------------------------------

__device__ __forceinline__ v8f wmma_f16(v16h a, v16h b, v8f c) {
  return __builtin_amdgcn_wmma_f32_16x16x32_f16(false, a, false, b, (short)0, c,
                                                false, false);
}

// A fragment (16x32 f16, row-major src, leading dim ld).
__device__ __forceinline__ v16h load_A(const _Float16* base, int ld, int mBase,
                                       int kBase) {
  const int lane = threadIdx.x & 31;
  const _Float16* p =
      base + (size_t)(mBase + (lane & 15)) * ld + kBase + ((lane >> 4) << 3);
  v8h lo = *(const v8h*)p;
  v8h hi = *(const v8h*)(p + 16);
  return __builtin_shufflevector(lo, hi, 0, 1, 2, 3, 4, 5, 6, 7, 8, 9, 10, 11,
                                 12, 13, 14, 15);
}

// B fragment for X @ W^T: B[k][n] = W[n][k]; W row-major [N,K], leading dim ld.
__device__ __forceinline__ v16h load_B(const _Float16* W, int ld, int nBase,
                                       int kBase) {
  const int lane = threadIdx.x & 31;
  const _Float16* p =
      W + (size_t)(nBase + (lane & 15)) * ld + kBase + ((lane >> 4) << 3);
  v8h lo = *(const v8h*)p;
  v8h hi = *(const v8h*)(p + 16);
  return __builtin_shufflevector(lo, hi, 0, 1, 2, 3, 4, 5, 6, 7, 8, 9, 10, 11,
                                 12, 13, 14, 15);
}

__device__ __forceinline__ float sigm_(float x) {
  return 1.0f / (1.0f + __expf(-x));
}
__device__ __forceinline__ float tanh_(float x) {
  x = fminf(fmaxf(x, -15.0f), 15.0f);
  float e = __expf(2.0f * x);
  return (e - 1.0f) / (e + 1.0f);
}

// ---- fp32 -> f16 staging ---------------------------------------------------

__global__ void cvt_f32_f16_kernel(const float* __restrict__ s,
                                   _Float16* __restrict__ d, int n) {
  for (int i = blockIdx.x * blockDim.x + threadIdx.x; i < n;
       i += gridDim.x * blockDim.x)
    d[i] = (_Float16)s[i];
}

// ---- value branch: relu(relu(obs@v1^T+b1)@v2^T+b2)@v3^T+b3 -----------------
// 16 WGs x 256 threads; 64 batch rows per WG.

__global__ __launch_bounds__(256) void value_kernel(
    const _Float16* __restrict__ obs_h, const _Float16* __restrict__ w1,
    const float* __restrict__ b1, const _Float16* __restrict__ w2,
    const float* __restrict__ b2, const float* __restrict__ w3,
    const float* __restrict__ b3, float* __restrict__ out) {
  __shared__ alignas(16) _Float16 act1[64 * 136];
  __shared__ float act2[64 * 132];

  const int tid = threadIdx.x;
  const int wave = tid >> 5;
  const int lane = tid & 31;
  const int rowBase = blockIdx.x * 64;
  const int mT = wave >> 1;              // 0..3
  const int nG = (wave & 1) * 4;         // 4 n-tiles per wave
  const int cRow = (lane >> 4) << 3;
  const int cCol = lane & 15;

  const _Float16* arow = obs_h + (size_t)rowBase * KFLAT;

  // GEMM1: [64,4096] x [4096 -> 128]
  v8f acc[4];
#pragma unroll
  for (int i = 0; i < 4; ++i) acc[i] = {};
  for (int k = 0; k < KFLAT; k += 32) {
    v16h a = load_A(arow, KFLAT, mT * 16, k);
#pragma unroll
    for (int i = 0; i < 4; ++i) {
      v16h b = load_B(w1, KFLAT, (nG + i) * 16, k);
      acc[i] = wmma_f16(a, b, acc[i]);
    }
  }
#pragma unroll
  for (int i = 0; i < 4; ++i) {
    int nb = (nG + i) * 16 + cCol;
#pragma unroll
    for (int e = 0; e < 8; ++e) {
      int r = mT * 16 + cRow + e;
      act1[r * 136 + nb] = (_Float16)fmaxf(acc[i][e] + b1[nb], 0.0f);
    }
  }
  __syncthreads();

  // GEMM2: [64,128] x [128 -> 128]
  v8f acc2[4];
#pragma unroll
  for (int i = 0; i < 4; ++i) acc2[i] = {};
#pragma unroll
  for (int k = 0; k < HID; k += 32) {
    v16h a = load_A(act1, 136, mT * 16, k);
#pragma unroll
    for (int i = 0; i < 4; ++i) {
      v16h b = load_B(w2, HID, (nG + i) * 16, k);
      acc2[i] = wmma_f16(a, b, acc2[i]);
    }
  }
#pragma unroll
  for (int i = 0; i < 4; ++i) {
    int nb = (nG + i) * 16 + cCol;
#pragma unroll
    for (int e = 0; e < 8; ++e) {
      int r = mT * 16 + cRow + e;
      act2[r * 132 + nb] = fmaxf(acc2[i][e] + b2[nb], 0.0f);
    }
  }
  __syncthreads();

  // GEMM3 (N=1): 4 lanes per row + shuffle reduce
  {
    int b = tid >> 2;   // 0..63
    int q = tid & 3;    // 0..3
    float s = 0.0f;
#pragma unroll
    for (int i = 0; i < 32; ++i) {
      int h = q * 32 + i;
      s += act2[b * 132 + h] * w3[h];
    }
    s += __shfl_xor(s, 1, 32);
    s += __shfl_xor(s, 2, 32);
    if (q == 0) out[(size_t)BATCH * 256 + rowBase + b] = s + b3[0];
  }
}

// ---- encoder: 128 sequential GRU steps, 16 batch rows per WG (64 WGs) ------

__global__ __launch_bounds__(256) void encoder_kernel(
    const _Float16* __restrict__ obs_h, const _Float16* __restrict__ wEmb,
    const float* __restrict__ bEmb, const _Float16* __restrict__ wIh,
    const _Float16* __restrict__ wHh, const float* __restrict__ bIh,
    const float* __restrict__ bHh, _Float16* __restrict__ encT,
    float* __restrict__ encH) {
  __shared__ float h32[16 * 132];
  __shared__ float grz[16 * 260];   // r:0..127, z:128..255 (post-sigmoid)
  __shared__ float gin[16 * 132];
  __shared__ float ghn[16 * 132];
  __shared__ alignas(16) _Float16 h16[16 * 136];
  __shared__ alignas(16) _Float16 e16[16 * 136];

  const int tid = threadIdx.x;
  const int wave = tid >> 5;
  const int lane = tid & 31;
  const int rowBase = blockIdx.x * 16;
  const int cRow = (lane >> 4) << 3;
  const int cCol = lane & 15;

  for (int i = tid; i < 16 * 132; i += 256) h32[i] = 0.0f;
  for (int i = tid; i < 16 * 136; i += 256) h16[i] = (_Float16)0.0f;
  __syncthreads();

  const _Float16* arow = obs_h + (size_t)rowBase * KFLAT;

  for (int t = 0; t < SEQ; ++t) {
    // e = x_t @ Wemb^T + bEmb    (8 n-tiles, one per wave, K=32)
    {
      int nBase = wave * 16;
      v8f acc = {};
      acc = wmma_f16(load_A(arow, KFLAT, 0, t * OBSZ),
                     load_B(wEmb, OBSZ, nBase, 0), acc);
      int nb = nBase + cCol;
#pragma unroll
      for (int e = 0; e < 8; ++e)
        e16[(cRow + e) * 136 + nb] = (_Float16)(acc[e] + bEmb[nb]);
    }
    __syncthreads();

    // gates: fused (r,z) 16 tiles; i_n 8 tiles; h_n 8 tiles
    for (int tile = wave; tile < 32; tile += 8) {
      if (tile < 16) {
        int nBase = tile * 16;
        v8f acc = {};
#pragma unroll
        for (int k = 0; k < HID; k += 32)
          acc = wmma_f16(load_A(e16, 136, 0, k), load_B(wIh, HID, nBase, k),
                         acc);
#pragma unroll
        for (int k = 0; k < HID; k += 32)
          acc = wmma_f16(load_A(h16, 136, 0, k), load_B(wHh, HID, nBase, k),
                         acc);
        int nb = nBase + cCol;
#pragma unroll
        for (int e = 0; e < 8; ++e)
          grz[(cRow + e) * 260 + nb] = sigm_(acc[e] + bIh[nb] + bHh[nb]);
      } else if (tile < 24) {  // i_n
        int nBase = (tile - 16) * 16;
        v8f acc = {};
#pragma unroll
        for (int k = 0; k < HID; k += 32)
          acc = wmma_f16(load_A(e16, 136, 0, k),
                         load_B(wIh, HID, 256 + nBase, k), acc);
        int nb = nBase + cCol;
#pragma unroll
        for (int e = 0; e < 8; ++e)
          gin[(cRow + e) * 132 + nb] = acc[e] + bIh[256 + nb];
      } else {  // h_n
        int nBase = (tile - 24) * 16;
        v8f acc = {};
#pragma unroll
        for (int k = 0; k < HID; k += 32)
          acc = wmma_f16(load_A(h16, 136, 0, k),
                         load_B(wHh, HID, 256 + nBase, k), acc);
        int nb = nBase + cCol;
#pragma unroll
        for (int e = 0; e < 8; ++e)
          ghn[(cRow + e) * 132 + nb] = acc[e] + bHh[256 + nb];
      }
    }
    __syncthreads();

    // h = (1-z)*tanh(i_n + r*h_n) + z*h
    for (int idx = tid; idx < 16 * HID; idx += 256) {
      int b = idx >> 7, j = idx & 127;
      float r = grz[b * 260 + j];
      float z = grz[b * 260 + 128 + j];
      float n = tanh_(gin[b * 132 + j] + r * ghn[b * 132 + j]);
      float hn = (1.0f - z) * n + z * h32[b * 132 + j];
      h32[b * 132 + j] = hn;
      h16[b * 136 + j] = (_Float16)hn;
    }
    // enc_outs = h2[batch row 0]; write transposed encT[h][t]
    if (rowBase == 0 && tid < HID) encT[tid * SEQ + t] = (_Float16)h32[tid];
    __syncthreads();
  }

  for (int idx = tid; idx < 16 * HID; idx += 256) {
    int b = idx >> 7, j = idx & 127;
    encH[(size_t)(rowBase + b) * HID + j] = h32[b * 132 + j];
  }
}

// ---- decoder: 128 autoregressive steps, 16 batch rows per WG (64 WGs) ------
// LDS: sg[] is scores during attn/softmax, then i_n during gates (disjoint).

__global__ __launch_bounds__(256) void decoder_kernel(
    const float* __restrict__ encH, const _Float16* __restrict__ encT,
    const float* __restrict__ wDemb, const float* __restrict__ bDemb,
    const _Float16* __restrict__ wAttn, const float* __restrict__ bAttn,
    const _Float16* __restrict__ wComb, const float* __restrict__ bComb,
    const _Float16* __restrict__ wIh, const _Float16* __restrict__ wHh,
    const float* __restrict__ bIh, const float* __restrict__ bHh,
    const float* __restrict__ wOut, const float* __restrict__ bOut,
    float* __restrict__ outBuf) {
  __shared__ float h32[16 * 132];
  __shared__ float sg[16 * 132];    // scores / i_n (lifetimes disjoint)
  __shared__ float grz[16 * 260];
  __shared__ float ghn[16 * 132];
  __shared__ float inp[16 * 2];
  __shared__ alignas(16) _Float16 h16[16 * 136];
  __shared__ alignas(16) _Float16 emb16[16 * 136];
  __shared__ alignas(16) _Float16 aw16[16 * 136];
  __shared__ alignas(16) _Float16 ctx16[16 * 136];
  __shared__ alignas(16) _Float16 o16[16 * 136];

  const int tid = threadIdx.x;
  const int wave = tid >> 5;
  const int lane = tid & 31;
  const int rowBase = blockIdx.x * 16;
  const int cRow = (lane >> 4) << 3;
  const int cCol = lane & 15;

  for (int idx = tid; idx < 16 * HID; idx += 256) {
    int b = idx >> 7, j = idx & 127;
    float v = encH[(size_t)(rowBase + b) * HID + j];
    h32[b * 132 + j] = v;
    h16[b * 136 + j] = (_Float16)v;
  }
  if (tid < 32) inp[tid] = 0.0f;
  __syncthreads();

  for (int t = 0; t < SEQ; ++t) {
    // emb = inp @ Wemb^T + b   (K=2, VALU)
    for (int idx = tid; idx < 16 * HID; idx += 256) {
      int b = idx >> 7, j = idx & 127;
      float v = inp[b * 2 + 0] * wDemb[j * 2 + 0] +
                inp[b * 2 + 1] * wDemb[j * 2 + 1] + bDemb[j];
      emb16[b * 136 + j] = (_Float16)v;
    }
    __syncthreads();

    // scores = concat(emb,h) @ attn_W^T + b   (8 n-tiles, 1/wave)
    {
      int nBase = wave * 16;
      v8f acc = {};
#pragma unroll
      for (int k = 0; k < HID; k += 32)
        acc = wmma_f16(load_A(emb16, 136, 0, k),
                       load_B(wAttn, 2 * HID, nBase, k), acc);
#pragma unroll
      for (int k = 0; k < HID; k += 32)
        acc = wmma_f16(load_A(h16, 136, 0, k),
                       load_B(wAttn, 2 * HID, nBase, HID + k), acc);
      int nb = nBase + cCol;
#pragma unroll
      for (int e = 0; e < 8; ++e)
        sg[(cRow + e) * 132 + nb] = acc[e] + bAttn[nb];
    }
    __syncthreads();

    // softmax: 8 lanes per row, shuffle-reduced
    if (tid < 128) {
      int b = tid >> 3, q = tid & 7;
      float loc[16];
      float m = -1e30f;
#pragma unroll
      for (int i = 0; i < 16; ++i) {
        loc[i] = sg[b * 132 + q * 16 + i];
        m = fmaxf(m, loc[i]);
      }
      m = fmaxf(m, __shfl_xor(m, 1, 32));
      m = fmaxf(m, __shfl_xor(m, 2, 32));
      m = fmaxf(m, __shfl_xor(m, 4, 32));
      float s = 0.0f;
#pragma unroll
      for (int i = 0; i < 16; ++i) {
        loc[i] = __expf(loc[i] - m);
        s += loc[i];
      }
      s += __shfl_xor(s, 1, 32);
      s += __shfl_xor(s, 2, 32);
      s += __shfl_xor(s, 4, 32);
      float inv = 1.0f / s;
#pragma unroll
      for (int i = 0; i < 16; ++i)
        aw16[b * 136 + q * 16 + i] = (_Float16)(loc[i] * inv);
    }
    __syncthreads();

    // ctx = aw @ enc_outs   (encT = enc_outs^T [H][L])
    {
      int nBase = wave * 16;
      v8f acc = {};
#pragma unroll
      for (int k = 0; k < SEQ; k += 32)
        acc = wmma_f16(load_A(aw16, 136, 0, k), load_B(encT, SEQ, nBase, k),
                       acc);
      int nb = nBase + cCol;
#pragma unroll
      for (int e = 0; e < 8; ++e)
        ctx16[(cRow + e) * 136 + nb] = (_Float16)acc[e];
    }
    __syncthreads();

    // o = tanh(concat(emb,ctx) @ comb_W^T + b)
    {
      int nBase = wave * 16;
      v8f acc = {};
#pragma unroll
      for (int k = 0; k < HID; k += 32)
        acc = wmma_f16(load_A(emb16, 136, 0, k),
                       load_B(wComb, 2 * HID, nBase, k), acc);
#pragma unroll
      for (int k = 0; k < HID; k += 32)
        acc = wmma_f16(load_A(ctx16, 136, 0, k),
                       load_B(wComb, 2 * HID, nBase, HID + k), acc);
      int nb = nBase + cCol;
#pragma unroll
      for (int e = 0; e < 8; ++e)
        o16[(cRow + e) * 136 + nb] = (_Float16)tanh_(acc[e] + bComb[nb]);
    }
    __syncthreads();

    // GRU gates with x = o
    for (int tile = wave; tile < 32; tile += 8) {
      if (tile < 16) {
        int nBase = tile * 16;
        v8f acc = {};
#pragma unroll
        for (int k = 0; k < HID; k += 32)
          acc = wmma_f16(load_A(o16, 136, 0, k), load_B(wIh, HID, nBase, k),
                         acc);
#pragma unroll
        for (int k = 0; k < HID; k += 32)
          acc = wmma_f16(load_A(h16, 136, 0, k), load_B(wHh, HID, nBase, k),
                         acc);
        int nb = nBase + cCol;
#pragma unroll
        for (int e = 0; e < 8; ++e)
          grz[(cRow + e) * 260 + nb] = sigm_(acc[e] + bIh[nb] + bHh[nb]);
      } else if (tile < 24) {  // i_n -> sg
        int nBase = (tile - 16) * 16;
        v8f acc = {};
#pragma unroll
        for (int k = 0; k < HID; k += 32)
          acc = wmma_f16(load_A(o16, 136, 0, k),
                         load_B(wIh, HID, 256 + nBase, k), acc);
        int nb = nBase + cCol;
#pragma unroll
        for (int e = 0; e < 8; ++e)
          sg[(cRow + e) * 132 + nb] = acc[e] + bIh[256 + nb];
      } else {  // h_n
        int nBase = (tile - 24) * 16;
        v8f acc = {};
#pragma unroll
        for (int k = 0; k < HID; k += 32)
          acc = wmma_f16(load_A(h16, 136, 0, k),
                         load_B(wHh, HID, 256 + nBase, k), acc);
        int nb = nBase + cCol;
#pragma unroll
        for (int e = 0; e < 8; ++e)
          ghn[(cRow + e) * 132 + nb] = acc[e] + bHh[256 + nb];
      }
    }
    __syncthreads();

    // h update
    for (int idx = tid; idx < 16 * HID; idx += 256) {
      int b = idx >> 7, j = idx & 127;
      float r = grz[b * 260 + j];
      float z = grz[b * 260 + 128 + j];
      float n = tanh_(sg[b * 132 + j] + r * ghn[b * 132 + j]);
      float hn = (1.0f - z) * n + z * h32[b * 132 + j];
      h32[b * 132 + j] = hn;
      h16[b * 136 + j] = (_Float16)hn;
    }
    __syncthreads();

    // out = tanh(h @ out_W^T + b): 8 lanes per (row, action)
    {
      int b = tid >> 4;        // 0..15
      int a = (tid >> 3) & 1;  // 0..1
      int q = tid & 7;         // 0..7
      float s = 0.0f;
#pragma unroll
      for (int i = 0; i < 16; ++i) {
        int h = q * 16 + i;
        s += h32[b * 132 + h] * wOut[a * HID + h];
      }
      s += __shfl_xor(s, 1, 32);
      s += __shfl_xor(s, 2, 32);
      s += __shfl_xor(s, 4, 32);
      if (q == 0) {
        s = tanh_(s + bOut[a]);
        outBuf[(size_t)(rowBase + b) * 256 + t * 2 + a] = s;
        inp[b * 2 + a] = s;
      }
    }
    __syncthreads();
  }
}

// ---------------------------------------------------------------------------

extern "C" void kernel_launch(void* const* d_in, const int* in_sizes, int n_in,
                              void* d_out, int out_size, void* d_ws,
                              size_t ws_size, hipStream_t stream) {
  (void)in_sizes; (void)n_in; (void)out_size; (void)ws_size;

  const float* obs       = (const float*)d_in[0];
  const float* enc_emb_W = (const float*)d_in[1];
  const float* enc_emb_b = (const float*)d_in[2];
  const float* enc_W_ih  = (const float*)d_in[3];
  const float* enc_W_hh  = (const float*)d_in[4];
  const float* enc_b_ih  = (const float*)d_in[5];
  const float* enc_b_hh  = (const float*)d_in[6];
  const float* dec_emb_W = (const float*)d_in[7];
  const float* dec_emb_b = (const float*)d_in[8];
  const float* attn_W    = (const float*)d_in[9];
  const float* attn_b    = (const float*)d_in[10];
  const float* comb_W    = (const float*)d_in[11];
  const float* comb_b    = (const float*)d_in[12];
  const float* dec_W_ih  = (const float*)d_in[13];
  const float* dec_W_hh  = (const float*)d_in[14];
  const float* dec_b_ih  = (const float*)d_in[15];
  const float* dec_b_hh  = (const float*)d_in[16];
  const float* out_W     = (const float*)d_in[17];
  const float* out_b     = (const float*)d_in[18];
  const float* v1_W      = (const float*)d_in[19];
  const float* v1_b      = (const float*)d_in[20];
  const float* v2_W      = (const float*)d_in[21];
  const float* v2_b      = (const float*)d_in[22];
  const float* v3_W      = (const float*)d_in[23];
  const float* v3_b      = (const float*)d_in[24];

  char* ws = (char*)d_ws;
  size_t off = 0;
  auto takeH = [&](size_t elems) {
    _Float16* p = (_Float16*)(ws + off);
    off += (elems * 2 + 255) & ~(size_t)255;
    return p;
  };
  _Float16* obs_h = takeH((size_t)BATCH * KFLAT);
  _Float16* encT  = takeH((size_t)HID * SEQ);
  _Float16* wEE   = takeH(HID * OBSZ);
  _Float16* wEIH  = takeH(3 * HID * HID);
  _Float16* wEHH  = takeH(3 * HID * HID);
  _Float16* wAT   = takeH(SEQ * 2 * HID);
  _Float16* wCB   = takeH(HID * 2 * HID);
  _Float16* wDIH  = takeH(3 * HID * HID);
  _Float16* wDHH  = takeH(3 * HID * HID);
  _Float16* wV1   = takeH((size_t)HID * KFLAT);
  _Float16* wV2   = takeH(HID * HID);
  float* encH = (float*)(ws + off);  // BATCH*HID f32

  // fp32 -> f16 staging
  cvt_f32_f16_kernel<<<1024, 256, 0, stream>>>(obs, obs_h, BATCH * KFLAT);
  cvt_f32_f16_kernel<<<16, 256, 0, stream>>>(enc_emb_W, wEE, HID * OBSZ);
  cvt_f32_f16_kernel<<<64, 256, 0, stream>>>(enc_W_ih, wEIH, 3 * HID * HID);
  cvt_f32_f16_kernel<<<64, 256, 0, stream>>>(enc_W_hh, wEHH, 3 * HID * HID);
  cvt_f32_f16_kernel<<<64, 256, 0, stream>>>(attn_W, wAT, SEQ * 2 * HID);
  cvt_f32_f16_kernel<<<64, 256, 0, stream>>>(comb_W, wCB, HID * 2 * HID);
  cvt_f32_f16_kernel<<<64, 256, 0, stream>>>(dec_W_ih, wDIH, 3 * HID * HID);
  cvt_f32_f16_kernel<<<64, 256, 0, stream>>>(dec_W_hh, wDHH, 3 * HID * HID);
  cvt_f32_f16_kernel<<<512, 256, 0, stream>>>(v1_W, wV1, HID * KFLAT);
  cvt_f32_f16_kernel<<<32, 256, 0, stream>>>(v2_W, wV2, HID * HID);

  value_kernel<<<BATCH / 64, 256, 0, stream>>>(obs_h, wV1, v1_b, wV2, v2_b,
                                               v3_W, v3_b, (float*)d_out);

  encoder_kernel<<<BATCH / 16, 256, 0, stream>>>(
      obs_h, wEE, enc_emb_b, wEIH, wEHH, enc_b_ih, enc_b_hh, encT, encH);

  decoder_kernel<<<BATCH / 16, 256, 0, stream>>>(
      encH, encT, dec_emb_W, dec_emb_b, wAT, attn_b, wCB, comb_b, wDIH, wDHH,
      dec_b_ih, dec_b_hh, out_W, out_b, (float*)d_out);
}
